// DecoderBlock_15874199126043
// MI455X (gfx1250) — matrix operand order repaired and checked by reading, not compiled, not run on hardware
//
#include <hip/hip_runtime.h>
#include <hip/hip_bf16.h>

// ---------------------------------------------------------------------------
// Types for CDNA5 WMMA (wave32): 16x16x32 bf16 -> f32
// ---------------------------------------------------------------------------
typedef __attribute__((ext_vector_type(8)))  __bf16 v8bf;
typedef __attribute__((ext_vector_type(16))) __bf16 v16bf;
typedef __attribute__((ext_vector_type(8)))  float  v8f;

#define B_SZ   4
#define T_SEQ  1024
#define E_DIM  1024
#define HEADS  16
#define HSZ    64
#define MT     (B_SZ * T_SEQ)          // 4096 rows of activations
#define E4     (4 * E_DIM)             // 4096 FFN hidden
#define BH     (B_SZ * HEADS)          // 64 attention batches
#define CHUNK  8                       // heads per attention chunk

// ---------------------------------------------------------------------------
// Tiled batched GEMM:  C[M,N] = A[M,K] (bf16, row major) x Bt[N,K]^T (bf16)
// Per block: 128x128 tile, K-step 64 (= 2 WMMA k-steps). 8 waves, each wave
// computes a 64x32 sub-tile as 4x2 WMMA 16x16 accumulators.
// Staging: CDNA5 async global->LDS (ASYNCcnt), double-buffered so the copy of
// tile k+1 overlaps the WMMA work of tile k with a single barrier per tile.
// IOFFSET is applied by HW to BOTH the global and LDS side, so the four 16B
// slices per thread share one address pair and differ only in offset: imm.
// NOTE: the shared array pointer is passed into the asm as a dummy operand so
// the compiler knows the asm writes LDS (otherwise the ds_load fragments get
// folded to undef, as observed in round 2's disassembly).
// ---------------------------------------------------------------------------
#define BLK_M 128
#define BLK_N 128
#define BLK_K 64
#define LDSB  144                      // LDS bytes per tile row: 128B data + 16B pad
#define BUFSZ (2 * BLK_M * LDSB)       // one double-buffer slot: A tile + B tile

#define ASYNC_CP(L, G, OFF)                                                   \
  asm volatile("global_load_async_to_lds_b128 %0, %1, off offset:" #OFF       \
               :: "v"(L), "v"(G), "v"((const unsigned char*)lds) : "memory")

__global__ __launch_bounds__(256)
void gemm_bf16_wmma(const __bf16* __restrict__ A, const __bf16* __restrict__ Bt,
                    const float* __restrict__ bias,
                    float* __restrict__ Cf, __bf16* __restrict__ Cb,
                    int M, int N, int K,
                    long long sA, long long sB, long long sC, int relu)
{
  __shared__ unsigned char lds[2 * BUFSZ];   // 72 KB of the 320 KB WGP LDS

  const int tid  = threadIdx.x;
  const int lane = tid & 31;          // wave32
  const int wid  = tid >> 5;          // 8 waves
  const int m0   = blockIdx.y * BLK_M;
  const int n0   = blockIdx.x * BLK_N;
  const int z    = blockIdx.z;

  A  += (size_t)z * sA;
  Bt += (size_t)z * sB;
  if (Cf) Cf += (size_t)z * sC;
  if (Cb) Cb += (size_t)z * sC;

  // staging: thread t covers 64B of row t/2 at K-chunk (t&1)*32 elements.
  // Async loads cannot zero-fill, so clamp ragged rows: the garbage only
  // reaches accumulator columns >= N which the epilogue never stores.
  const int srow  = tid >> 1;
  const int skel  = (tid & 1) * 32;   // element offset within K tile
  const int sbyte = (tid & 1) * 64;   // byte offset within LDS row
  const int arow  = (m0 + srow < M) ? (m0 + srow) : (M - 1);
  const int brow  = (n0 + srow < N) ? (n0 + srow) : (N - 1);

  const unsigned long long gA = (unsigned long long)(A  + (size_t)arow * K + skel);
  const unsigned long long gB = (unsigned long long)(Bt + (size_t)brow * K + skel);
  const unsigned ldsAoff = (unsigned)(srow * LDSB + sbyte);
  const unsigned ldsBoff = (unsigned)(BLK_M * LDSB + srow * LDSB + sbyte);

  // issue 8 async 16B copies (4 for A tile, 4 for B tile) into buffer `buf`
  auto issue_tile = [&](int buf, int k0) {
    const unsigned a3 = (unsigned)(buf * BUFSZ) + ldsAoff;
    const unsigned b3 = (unsigned)(buf * BUFSZ) + ldsBoff;
    const unsigned long long ga = gA + (unsigned long long)k0 * 2;  // bf16 bytes
    const unsigned long long gb = gB + (unsigned long long)k0 * 2;
    ASYNC_CP(a3, ga, 0);   ASYNC_CP(b3, gb, 0);
    ASYNC_CP(a3, ga, 16);  ASYNC_CP(b3, gb, 16);
    ASYNC_CP(a3, ga, 32);  ASYNC_CP(b3, gb, 32);
    ASYNC_CP(a3, ga, 48);  ASYNC_CP(b3, gb, 48);
  };

  // wave sub-tile origin inside the 128x128 block tile
  const int m_w = (wid >> 2) * 64;    // 2 waves along M
  const int n_w = (wid & 3) * 32;     // 4 waves along N
  const int l16 = lane & 15;
  const int lh  = lane >> 4;          // lane half selects K sub-chunk

  v8f acc[4][2];
  #pragma unroll
  for (int i = 0; i < 4; i++)
    #pragma unroll
    for (int j = 0; j < 2; j++)
      acc[i][j] = (v8f){0.f,0.f,0.f,0.f,0.f,0.f,0.f,0.f};

  // ---- async double-buffered main loop -----------------------------------
  int buf = 0;
  issue_tile(0, 0);
  asm volatile("s_wait_asynccnt 0" ::: "memory");
  __syncthreads();

  for (int k0 = 0; k0 < K; k0 += BLK_K) {
    if (k0 + BLK_K < K)
      issue_tile(buf ^ 1, k0 + BLK_K);   // overlap next copy with this compute

    const unsigned char* ldsA = lds + buf * BUFSZ;
    const unsigned char* ldsB = ldsA + BLK_M * LDSB;

    #pragma unroll
    for (int ks = 0; ks < 2; ks++) {   // two 16x16x32 k-steps per 64-wide stage
      v16bf af[4], bfr[2];
      #pragma unroll
      for (int mt = 0; mt < 4; mt++) { // A frag: lane row = m, K chunks 8 + 8
        const unsigned char* p = ldsA + (m_w + mt * 16 + l16) * LDSB + ks * 64 + lh * 16;
        v8bf lo = *(const v8bf*)p;
        v8bf hi = *(const v8bf*)(p + 32);
        af[mt] = __builtin_shufflevector(lo, hi, 0,1,2,3,4,5,6,7,8,9,10,11,12,13,14,15);
      }
      #pragma unroll
      for (int nt = 0; nt < 2; nt++) { // B frag: lane col = n, K chunks 8 + 8
        const unsigned char* p = ldsB + (n_w + nt * 16 + l16) * LDSB + ks * 64 + lh * 16;
        v8bf lo = *(const v8bf*)p;
        v8bf hi = *(const v8bf*)(p + 32);
        bfr[nt] = __builtin_shufflevector(lo, hi, 0,1,2,3,4,5,6,7,8,9,10,11,12,13,14,15);
      }
      #pragma unroll
      for (int mt = 0; mt < 4; mt++)
        #pragma unroll
        for (int nt = 0; nt < 2; nt++)
          acc[mt][nt] = __builtin_amdgcn_wmma_f32_16x16x32_bf16(
              false, af[mt], false, bfr[nt], (short)0, acc[mt][nt], false, false);
    }

    // next buffer's async copies complete + everyone done reading this buffer
    asm volatile("s_wait_asynccnt 0" ::: "memory");
    __syncthreads();
    buf ^= 1;
  }

  // Epilogue. C/D layout: VGPR v, lane L -> row = v + (L>>4)*8, col = L&15.
  #pragma unroll
  for (int nt = 0; nt < 2; nt++) {
    const int col = n0 + n_w + nt * 16 + l16;
    if (col >= N) continue;
    const float bv = bias ? bias[col] : 0.f;
    #pragma unroll
    for (int mt = 0; mt < 4; mt++) {
      const int rbase = m0 + m_w + mt * 16 + lh * 8;
      #pragma unroll
      for (int v = 0; v < 8; v++) {
        const int row = rbase + v;
        if (row >= M) continue;
        float x = acc[mt][nt][v] + bv;
        if (relu) x = fmaxf(x, 0.f);
        const size_t o = (size_t)row * N + col;
        if (Cf) Cf[o] = x;
        if (Cb) Cb[o] = (__bf16)x;
      }
    }
  }
}

// ---------------------------------------------------------------------------
// Elementwise converters / reshapes
// ---------------------------------------------------------------------------
__global__ void cvt_f32_bf16(const float* __restrict__ src, __bf16* __restrict__ dst, size_t n)
{
  size_t i = (size_t)blockIdx.x * 256 + threadIdx.x;
  if (i < n) dst[i] = (__bf16)src[i];
}

// src [rows][cols] fp32 -> dst [cols][rows] bf16  (weight [in][out] -> [out][in])
__global__ void cvt_transpose_bf16(const float* __restrict__ src, __bf16* __restrict__ dst,
                                   int rows, int cols)
{
  size_t i = (size_t)blockIdx.x * 256 + threadIdx.x;
  if (i >= (size_t)rows * cols) return;
  int c = (int)(i % cols);
  int r = (int)(i / cols);
  dst[(size_t)c * rows + r] = (__bf16)src[i];
}

// [B,T,H,HS] fp32 -> bf16 head-major [BH,T,HS] (transpose=0) or [BH,HS,T] (=1, for V)
__global__ void pack_heads(const float* __restrict__ src, __bf16* __restrict__ dst, int transpose)
{
  size_t i = (size_t)blockIdx.x * 256 + threadIdx.x;
  if (i >= (size_t)MT * E_DIM) return;
  int d = (int)(i % HSZ);
  size_t r = i / HSZ;
  int h = (int)(r % HEADS); r /= HEADS;
  int t = (int)(r % T_SEQ);
  int b = (int)(r / T_SEQ);
  float v = src[i];
  if (transpose)
    dst[((size_t)(b * HEADS + h) * HSZ + d) * T_SEQ + t] = (__bf16)v;
  else
    dst[((size_t)(b * HEADS + h) * T_SEQ + t) * HSZ + d] = (__bf16)v;
}

// [BH,T,HS] fp32 -> [B,T,E] bf16 (concat heads for output projection)
__global__ void unpack_heads(const float* __restrict__ src, __bf16* __restrict__ dst)
{
  size_t i = (size_t)blockIdx.x * 256 + threadIdx.x;
  if (i >= (size_t)MT * E_DIM) return;
  int d = (int)(i % HSZ);
  size_t r = i / HSZ;
  int t = (int)(r % T_SEQ); r /= T_SEQ;
  int h = (int)(r % HEADS);
  int b = (int)(r / HEADS);
  dst[((size_t)(b * T_SEQ + t)) * E_DIM + h * HSZ + d] = (__bf16)src[i];
}

// ---------------------------------------------------------------------------
// Row softmax over T=1024 keys. Mask (key < query) applied before 1/sqrt(HS),
// matching the reference. Emits bf16 probabilities for the AV WMMA GEMM.
// ---------------------------------------------------------------------------
__global__ __launch_bounds__(256)
void softmax_rows(const float* __restrict__ S, __bf16* __restrict__ P, int causal)
{
  __shared__ float red[256];
  const int tid = threadIdx.x;
  const size_t row = blockIdx.x;
  const int q = (int)(row % T_SEQ);
  const float* s = S + row * T_SEQ;
  __bf16* p = P + row * T_SEQ;

  float vals[4];
  float mx = -3.4e38f;
  #pragma unroll
  for (int i = 0; i < 4; i++) {
    int j = tid + i * 256;
    float x = s[j];
    if (causal && (j < q)) x = -1.0e9f;   // torch: mask BEFORE scaling
    x *= 0.125f;                           // 1/sqrt(64)
    vals[i] = x;
    mx = fmaxf(mx, x);
  }
  red[tid] = mx; __syncthreads();
  for (int st = 128; st > 0; st >>= 1) { if (tid < st) red[tid] = fmaxf(red[tid], red[tid + st]); __syncthreads(); }
  mx = red[0]; __syncthreads();

  float sum = 0.f;
  #pragma unroll
  for (int i = 0; i < 4; i++) { float e = __expf(vals[i] - mx); vals[i] = e; sum += e; }
  red[tid] = sum; __syncthreads();
  for (int st = 128; st > 0; st >>= 1) { if (tid < st) red[tid] += red[tid + st]; __syncthreads(); }
  const float rinv = 1.f / red[0];

  #pragma unroll
  for (int i = 0; i < 4; i++) { int j = tid + i * 256; p[j] = (__bf16)(vals[i] * rinv); }
}

// ---------------------------------------------------------------------------
// y = LayerNorm(x + r) * gamma + beta ; fp32 out + optional bf16 out
// ---------------------------------------------------------------------------
__global__ __launch_bounds__(256)
void add_layernorm(const float* __restrict__ X, const float* __restrict__ R,
                   const float* __restrict__ g, const float* __restrict__ b,
                   float* __restrict__ outF, __bf16* __restrict__ outB)
{
  __shared__ float red[256];
  const int tid = threadIdx.x;
  const size_t row = blockIdx.x;
  const float* x = X + row * E_DIM;
  const float* r = R + row * E_DIM;

  float v[4]; float s = 0.f;
  #pragma unroll
  for (int i = 0; i < 4; i++) { int j = tid + i * 256; v[i] = x[j] + r[j]; s += v[i]; }
  red[tid] = s; __syncthreads();
  for (int st = 128; st > 0; st >>= 1) { if (tid < st) red[tid] += red[tid + st]; __syncthreads(); }
  const float mu = red[0] * (1.f / E_DIM); __syncthreads();

  s = 0.f;
  #pragma unroll
  for (int i = 0; i < 4; i++) { float d = v[i] - mu; s += d * d; }
  red[tid] = s; __syncthreads();
  for (int st = 128; st > 0; st >>= 1) { if (tid < st) red[tid] += red[tid + st]; __syncthreads(); }
  const float inv = rsqrtf(red[0] * (1.f / E_DIM) + 1e-5f);

  #pragma unroll
  for (int i = 0; i < 4; i++) {
    int j = tid + i * 256;
    float y = (v[i] - mu) * inv * g[j] + b[j];
    if (outF) outF[row * E_DIM + j] = y;
    if (outB) outB[row * E_DIM + j] = (__bf16)y;
  }
}

// ---------------------------------------------------------------------------
// Host orchestration
// ---------------------------------------------------------------------------
extern "C" void kernel_launch(void* const* d_in, const int* in_sizes, int n_in,
                              void* d_out, int out_size, void* d_ws, size_t ws_size,
                              hipStream_t stream)
{
  (void)in_sizes; (void)n_in; (void)out_size; (void)ws_size;

  const float* target = (const float*)d_in[0];
  const float* enc    = (const float*)d_in[1];
  const float* sa_w[4] = { (const float*)d_in[2], (const float*)d_in[4], (const float*)d_in[6], (const float*)d_in[8] };
  const float* sa_b[4] = { (const float*)d_in[3], (const float*)d_in[5], (const float*)d_in[7], (const float*)d_in[9] };
  const float* ca_w[4] = { (const float*)d_in[10], (const float*)d_in[12], (const float*)d_in[14], (const float*)d_in[16] };
  const float* ca_b[4] = { (const float*)d_in[11], (const float*)d_in[13], (const float*)d_in[15], (const float*)d_in[17] };
  const float* ln1_g = (const float*)d_in[18]; const float* ln1_b = (const float*)d_in[19];
  const float* ln2_g = (const float*)d_in[20]; const float* ln2_b = (const float*)d_in[21];
  const float* ln3_g = (const float*)d_in[22]; const float* ln3_b = (const float*)d_in[23];
  const float* ff_w1 = (const float*)d_in[24]; const float* ff_b1 = (const float*)d_in[25];
  const float* ff_w2 = (const float*)d_in[26]; const float* ff_b2 = (const float*)d_in[27];

  char* ws = (char*)d_ws;
  size_t off = 0;
  auto alloc = [&](size_t bytes) -> void* {
    off = (off + 255) & ~(size_t)255;
    void* p = ws + off;
    off += bytes;
    return p;
  };

  // bf16 transposed weights [out][in]
  __bf16* wT[8];
  for (int i = 0; i < 8; i++) wT[i] = (__bf16*)alloc((size_t)E_DIM * E_DIM * 2);
  __bf16* ff1T = (__bf16*)alloc((size_t)E4 * E_DIM * 2);   // [4E, E]
  __bf16* ff2T = (__bf16*)alloc((size_t)E_DIM * E4 * 2);   // [E, 4E]

  __bf16* Xbf = (__bf16*)alloc((size_t)MT * E_DIM * 2);    // target bf16
  __bf16* Ebf = (__bf16*)alloc((size_t)MT * E_DIM * 2);    // enc_src bf16

  float*  Qf  = (float*)alloc((size_t)MT * E_DIM * 4);
  float*  Kf  = (float*)alloc((size_t)MT * E_DIM * 4);
  float*  Vf  = (float*)alloc((size_t)MT * E_DIM * 4);
  __bf16* Qh  = (__bf16*)alloc((size_t)BH * T_SEQ * HSZ * 2);   // [BH,T,HS]
  __bf16* Kh  = (__bf16*)alloc((size_t)BH * T_SEQ * HSZ * 2);   // [BH,T,HS]
  __bf16* Vh  = (__bf16*)alloc((size_t)BH * HSZ * T_SEQ * 2);   // [BH,HS,T]
  float*  Sc  = (float*)alloc((size_t)CHUNK * T_SEQ * T_SEQ * 4);
  __bf16* Pb  = (__bf16*)alloc((size_t)CHUNK * T_SEQ * T_SEQ * 2);
  float*  AVf = (float*)alloc((size_t)BH * T_SEQ * HSZ * 4);
  __bf16* Cc  = (__bf16*)alloc((size_t)MT * E_DIM * 2);
  float*  Pj  = (float*)alloc((size_t)MT * E_DIM * 4);
  float*  V1  = (float*)alloc((size_t)MT * E_DIM * 4);
  float*  B2  = (float*)alloc((size_t)MT * E_DIM * 4);
  __bf16* B2b = (__bf16*)alloc((size_t)MT * E_DIM * 2);
  __bf16* H1  = (__bf16*)alloc((size_t)MT * E4 * 2);
  float*  N2  = (float*)alloc((size_t)MT * E_DIM * 4);

  const size_t nAct = (size_t)MT * E_DIM;
  const unsigned actBlocks = (unsigned)((nAct + 255) / 256);

  auto gemm = [&](const __bf16* A, const __bf16* Bt, const float* bias,
                  float* Cf, __bf16* Cb, int M, int N, int K,
                  long long sA, long long sB, long long sC, int batches, int relu) {
    dim3 grid((N + BLK_N - 1) / BLK_N, (M + BLK_M - 1) / BLK_M, batches);
    gemm_bf16_wmma<<<grid, 256, 0, stream>>>(A, Bt, bias, Cf, Cb, M, N, K, sA, sB, sC, relu);
  };

  // ---- precision conversion of operands (once per call; deterministic) ----
  const unsigned wBlocks = (unsigned)(((size_t)E_DIM * E_DIM + 255) / 256);
  for (int i = 0; i < 4; i++)
    cvt_transpose_bf16<<<wBlocks, 256, 0, stream>>>(sa_w[i], wT[i], E_DIM, E_DIM);
  for (int i = 0; i < 4; i++)
    cvt_transpose_bf16<<<wBlocks, 256, 0, stream>>>(ca_w[i], wT[4 + i], E_DIM, E_DIM);
  cvt_transpose_bf16<<<(unsigned)(((size_t)E_DIM * E4 + 255) / 256), 256, 0, stream>>>(ff_w1, ff1T, E_DIM, E4);
  cvt_transpose_bf16<<<(unsigned)(((size_t)E4 * E_DIM + 255) / 256), 256, 0, stream>>>(ff_w2, ff2T, E4, E_DIM);
  cvt_f32_bf16<<<actBlocks, 256, 0, stream>>>(target, Xbf, nAct);
  cvt_f32_bf16<<<actBlocks, 256, 0, stream>>>(enc, Ebf, nAct);

  auto attention = [&](const __bf16* srcQ, const __bf16* srcKV,
                       __bf16* const* w, const float* const* bvec,
                       int causal, float* proj_out) {
    gemm(srcQ,  w[0], bvec[0], Qf, nullptr, MT, E_DIM, E_DIM, 0, 0, 0, 1, 0);
    gemm(srcKV, w[1], bvec[1], Kf, nullptr, MT, E_DIM, E_DIM, 0, 0, 0, 1, 0);
    gemm(srcKV, w[2], bvec[2], Vf, nullptr, MT, E_DIM, E_DIM, 0, 0, 0, 1, 0);
    pack_heads<<<actBlocks, 256, 0, stream>>>(Qf, Qh, 0);
    pack_heads<<<actBlocks, 256, 0, stream>>>(Kf, Kh, 0);
    pack_heads<<<actBlocks, 256, 0, stream>>>(Vf, Vh, 1);
    for (int c = 0; c < BH; c += CHUNK) {
      const __bf16* Qc = Qh + (size_t)c * T_SEQ * HSZ;
      const __bf16* Kc = Kh + (size_t)c * T_SEQ * HSZ;
      const __bf16* Vc = Vh + (size_t)c * HSZ * T_SEQ;
      float* AVc = AVf + (size_t)c * T_SEQ * HSZ;
      // scores = Q K^T  (A=[T,HS], Bt = K [keys][HS] already in [N][K] form)
      gemm(Qc, Kc, nullptr, Sc, nullptr, T_SEQ, T_SEQ, HSZ,
           (long long)T_SEQ * HSZ, (long long)T_SEQ * HSZ, (long long)T_SEQ * T_SEQ, CHUNK, 0);
      softmax_rows<<<CHUNK * T_SEQ, 256, 0, stream>>>(Sc, Pb, causal);
      // out = P V  (A=[T,T] probs, Bt = V^T [HS][T] in [N][K] form)
      gemm(Pb, Vc, nullptr, AVc, nullptr, T_SEQ, HSZ, T_SEQ,
           (long long)T_SEQ * T_SEQ, (long long)HSZ * T_SEQ, (long long)T_SEQ * HSZ, CHUNK, 0);
    }
    unpack_heads<<<actBlocks, 256, 0, stream>>>(AVf, Cc);
    gemm(Cc, w[3], bvec[3], proj_out, nullptr, MT, E_DIM, E_DIM, 0, 0, 0, 1, 0);
  };

  // ---- self attention + LN1 ----
  { __bf16* w[4] = { wT[0], wT[1], wT[2], wT[3] };
    const float* bv[4] = { sa_b[0], sa_b[1], sa_b[2], sa_b[3] };
    attention(Xbf, Xbf, w, bv, /*causal=*/1, Pj); }
  add_layernorm<<<MT, 256, 0, stream>>>(target, Pj, ln1_g, ln1_b, V1, nullptr);

  // ---- cross attention (query = target, faithful to reference) + LN2 ----
  { __bf16* w[4] = { wT[4], wT[5], wT[6], wT[7] };
    const float* bv[4] = { ca_b[0], ca_b[1], ca_b[2], ca_b[3] };
    attention(Xbf, Ebf, w, bv, /*causal=*/0, Pj); }
  add_layernorm<<<MT, 256, 0, stream>>>(V1, Pj, ln2_g, ln2_b, B2, B2b);

  // ---- FFN (ReLU fused in GEMM1, bf16-only intermediate) + LN3 -> d_out ----
  gemm(B2b, ff1T, ff_b1, nullptr, H1, MT, E4, E_DIM, 0, 0, 0, 1, /*relu=*/1);
  gemm(H1, ff2T, ff_b2, N2, nullptr, MT, E_DIM, E4, 0, 0, 0, 1, 0);
  add_layernorm<<<MT, 256, 0, stream>>>(B2, N2, ln3_g, ln3_b, (float*)d_out, nullptr);
}
